// Model_82600811036941
// MI455X (gfx1250) — matrix operand order repaired and checked by reading, not compiled, not run on hardware
//
#include <hip/hip_runtime.h>
#include <hip/hip_bf16.h>
#include <stdint.h>

typedef __attribute__((ext_vector_type(16))) _Float16 v16h;
typedef __attribute__((ext_vector_type(8)))  float    v8f;

#define BATCH  4096
#define TLEN   256
#define SLOTS  5
#define DIM    32
#define NTYPES 4
#define NJ     (NTYPES * SLOTS)   // 20 valid (type,slot) columns
#define LGW    32                 // logits LDS row stride (padded -> unconditional stores)

__device__ __forceinline__ float sigmoidf_(float x) {
    return 1.0f / (1.0f + __expf(-x));
}

// One block (256 threads = 8 wave32) per batch element.
__global__ __launch_bounds__(256)
void user_mem_net_kernel(const float* __restrict__ b_seq_emb,   // [B,T,32]
                         const float* __restrict__ q_emb,       // [B,32]
                         const float* __restrict__ keys,        // [4,5,32] == [20,32]
                         const float* __restrict__ Wu,          // [32,32]
                         const float* __restrict__ bu,          // [32]
                         const int*   __restrict__ b_seq_lens,  // [B]
                         const int*   __restrict__ b_seq_labels,// [B,T]
                         const int*   __restrict__ q_label,     // [B]
                         float* __restrict__ out)               // [B,32]
{
    // X stage (f32) and padded logits are both 256*32 floats: union them.
    __shared__ union {
        float Xf[TLEN * DIM];     // 32 KB  async-staged f32 sequence
        float lg[TLEN * LGW];     // 32 KB  logits, 32 padded columns
    } u;
    __shared__ float aat[TLEN * SLOTS];   // 5 KB  softmax a, then weights w
    __shared__ float vcoef[TLEN];         // 1 KB  per-step fold coefficient
    __shared__ float rattn[SLOTS];
    __shared__ float outacc[DIM];

    const int b    = blockIdx.x;
    const int tid  = threadIdx.x;
    const int lane = tid & 31;               // gfx1250 wave32
    const int wid  = tid >> 5;               // 0..7
    const int len  = b_seq_lens[b];
    const int* labels = b_seq_labels + b * TLEN;
    const float* Xg   = b_seq_emb + (size_t)b * TLEN * DIM;

    // ---- stage X into LDS via the async HBM->LDS engine (ASYNCcnt-tracked).
    // GVS mode: saddr = uniform base, vaddr = per-lane byte offset, vdst = LDS byte addr.
    {
        uint32_t ldsbase = (uint32_t)(uintptr_t)(&u.Xf[0]);   // low 32 bits = LDS address
        #pragma unroll
        for (int i = 0; i < 8; ++i) {                          // 2048 x 16B per block
            uint32_t off = (uint32_t)((tid + i * 256) * 16);
            uint32_t lds = ldsbase + off;
            asm volatile("global_load_async_to_lds_b128 %0, %1, %2"
                         :: "v"(lds), "v"(off), "s"(Xg) : "memory");
        }
    }

    // ---- constant B fragments: Wu (two N-tiles) and keys^T (cols 0..19, pad to 32)
    // B 32x16 f16 layout: col n = lane&15; lanes<16: K in {0..7,16..23}; lanes>=16: {8..15,24..31}
    const int ncol = lane & 15;
    const int kbB  = (lane < 16) ? 0 : 8;
    v16h bWu0, bWu1, bK0, bK1;
    #pragma unroll
    for (int i = 0; i < 16; ++i) {
        int k = (i < 8) ? (kbB + i) : (16 + kbB + (i - 8));
        bWu0[i] = (_Float16)Wu[k * DIM + ncol];
        bWu1[i] = (_Float16)Wu[k * DIM + (16 + ncol)];
        bK0[i]  = (_Float16)keys[ncol * DIM + k];               // j = ncol < 20 always
        int j1  = 16 + ncol;
        bK1[i]  = (j1 < NJ) ? (_Float16)keys[j1 * DIM + k] : (_Float16)0.0f;
    }

    asm volatile("s_wait_asynccnt 0" ::: "memory");
    __syncthreads();

    // ---- A fragments for this wave's two M-tiles (f32->f16 at fragment build;
    //      kept live in regs: reused by both GEMMs)
    // A 16x32 f16 layout: row m = lane&15; lanes<16: K in {0..7,16..23}; lanes>=16: {8..15,24..31}
    const int kbA = (lane < 16) ? 0 : 8;
    const int tb0 = wid * 16, tb1 = (wid + 8) * 16;
    v16h aFrag[2];
    {
        int tb[2] = { tb0, tb1 };
        #pragma unroll
        for (int mi = 0; mi < 2; ++mi) {
            int t = tb[mi] + (lane & 15);
            #pragma unroll
            for (int i = 0; i < 16; ++i) {
                int k = (i < 8) ? (kbA + i) : (16 + kbA + (i - 8));
                aFrag[mi][i] = (_Float16)u.Xf[t * DIM + k];
            }
        }
    }
    __syncthreads();   // Xf dead after this point; LDS reused as logits

    // ---- GEMM 1: logits = X @ keys^T  (one WMMA per 16x16 tile, K=32 exact)
    {
        int tb[2] = { tb0, tb1 };
        int mbase = (lane < 16) ? 0 : 8;    // C 16x16 f32: VGPR r -> row mbase+r, col ncol
        #pragma unroll
        for (int mi = 0; mi < 2; ++mi) {
            v8f c0 = {}, c1 = {};
            c0 = __builtin_amdgcn_wmma_f32_16x16x32_f16(false, aFrag[mi], false, bK0,
                                                        (short)0, c0, false, false);
            c1 = __builtin_amdgcn_wmma_f32_16x16x32_f16(false, aFrag[mi], false, bK1,
                                                        (short)0, c1, false, false);
            #pragma unroll
            for (int r = 0; r < 8; ++r) {
                int t = tb[mi] + mbase + r;
                u.lg[t * LGW + ncol]      = c0[r];   // unconditional: padded stride
                u.lg[t * LGW + 16 + ncol] = c1[r];
            }
        }
    }
    __syncthreads();

    // ---- per-row softmax over the 5 slots of eff_label (label if t < len-1 else 0)
    {
        int t = tid;                                   // 256 threads == TLEN rows
        int lab = (t < len - 1) ? labels[t] : 0;
        const float* row = &u.lg[t * LGW + lab * SLOTS];
        float mx = row[0];
        #pragma unroll
        for (int s = 1; s < SLOTS; ++s) mx = fmaxf(mx, row[s]);
        float e[SLOTS], den = 0.0f;
        #pragma unroll
        for (int s = 0; s < SLOTS; ++s) { e[s] = __expf(row[s] - mx); den += e[s]; }
        float inv = 1.0f / den;
        #pragma unroll
        for (int s = 0; s < SLOTS; ++s) aat[t * SLOTS + s] = e[s] * inv;
    }
    __syncthreads();

    // ---- closed-form scan: suffix products -> w[t,s] = valid * a*Π(1-a); r_attn; zero acc
    if (tid < SLOTS) {
        float P = 1.0f;
        for (int t = TLEN - 1; t >= 0; --t) {
            float a = aat[t * SLOTS + tid];
            if (t < len) { aat[t * SLOTS + tid] = a * P; P *= (1.0f - a); }
            else          aat[t * SLOTS + tid] = 0.0f;
        }
    } else if (tid == 8) {
        int ql = q_label[b];
        const float* kq = keys + (size_t)ql * SLOTS * DIM;
        const float* q  = q_emb + (size_t)b * DIM;
        float lq[SLOTS]; float mx = -3.4e38f;
        for (int s = 0; s < SLOTS; ++s) {
            float acc = 0.0f;
            for (int d = 0; d < DIM; ++d) acc += q[d] * kq[s * DIM + d];
            lq[s] = acc; mx = fmaxf(mx, acc);
        }
        float den = 0.0f;
        for (int s = 0; s < SLOTS; ++s) { lq[s] = __expf(lq[s] - mx); den += lq[s]; }
        float inv = 1.0f / den;
        for (int s = 0; s < SLOTS; ++s) rattn[s] = lq[s] * inv;
    } else if (tid >= 32 && tid < 32 + DIM) {
        outacc[tid - 32] = 0.0f;
    }
    __syncthreads();

    // ---- v[t] = Σ_s r_attn[s] * w[t,s]
    {
        int t = tid;
        float v = 0.0f;
        #pragma unroll
        for (int s = 0; s < SLOTS; ++s) v += rattn[s] * aat[t * SLOTS + s];
        vcoef[t] = v;
    }
    __syncthreads();

    // ---- GEMM 2: upd = sigmoid(X@Wu + bu), folded as out += v[t]*upd[t,:] (never stored)
    {
        float bu0 = bu[ncol], bu1 = bu[16 + ncol];
        int tb[2] = { tb0, tb1 };
        int mbase = (lane < 16) ? 0 : 8;
        #pragma unroll
        for (int mi = 0; mi < 2; ++mi) {
            v8f c0 = {}, c1 = {};
            c0 = __builtin_amdgcn_wmma_f32_16x16x32_f16(false, aFrag[mi], false, bWu0,
                                                        (short)0, c0, false, false);
            c1 = __builtin_amdgcn_wmma_f32_16x16x32_f16(false, aFrag[mi], false, bWu1,
                                                        (short)0, c1, false, false);
            float p0 = 0.0f, p1 = 0.0f;
            #pragma unroll
            for (int r = 0; r < 8; ++r) {
                int t = tb[mi] + mbase + r;
                float v = vcoef[t];
                p0 += v * sigmoidf_(c0[r] + bu0);
                p1 += v * sigmoidf_(c1[r] + bu1);
            }
            atomicAdd(&outacc[ncol], p0);        // ds_add_f32
            atomicAdd(&outacc[16 + ncol], p1);
        }
    }
    __syncthreads();

    if (tid < DIM) out[(size_t)b * DIM + tid] = outacc[tid];
}

extern "C" void kernel_launch(void* const* d_in, const int* in_sizes, int n_in,
                              void* d_out, int out_size, void* d_ws, size_t ws_size,
                              hipStream_t stream) {
    (void)in_sizes; (void)n_in; (void)out_size; (void)d_ws; (void)ws_size;
    const float* b_seq_emb    = (const float*)d_in[0];
    const float* q_emb        = (const float*)d_in[1];
    const float* keys         = (const float*)d_in[2];
    const float* Wu           = (const float*)d_in[3];
    const float* bu           = (const float*)d_in[4];
    const int*   b_seq_lens   = (const int*)  d_in[5];
    const int*   b_seq_labels = (const int*)  d_in[6];
    const int*   q_label      = (const int*)  d_in[7];
    float* out = (float*)d_out;

    user_mem_net_kernel<<<BATCH, 256, 0, stream>>>(
        b_seq_emb, q_emb, keys, Wu, bu, b_seq_lens, b_seq_labels, q_label, out);
}
